// VectorQuantizer_15135464751482
// MI455X (gfx1250) — matrix operand order repaired and checked by reading, not compiled, not run on hardware
//
#include <hip/hip_runtime.h>
#include <hip/hip_bf16.h>
#include <stddef.h>
#include <stdint.h>

// ---- CDNA5 (gfx1250) VQ-VAE vector quantizer -------------------------------
// dist argmin via v_wmma_f32_16x16x32_bf16 (f32 accumulate), B tiles staged in
// LDS with global_load_async_to_lds_b128 + s_wait_asynccnt double buffering.

typedef __bf16 bf16_t;
typedef __attribute__((ext_vector_type(16))) __bf16 v16bf;
typedef __attribute__((ext_vector_type(8)))  __bf16 v8bf;
typedef __attribute__((ext_vector_type(8)))  float  v8f;

#define K_CODES 1024
#define D_DIM   256
#define B_SZ    32
#define HW      1024                  // 32*32
#define N_TOT   (B_SZ * HW)           // 32768
#define NUMEL   ((size_t)N_TOT * D_DIM)
#define NKB     (K_CODES / 16)        // 64 code blocks
#define BPITCH  264                   // LDS row pitch (bf16): 528B = 132 dw -> bank stride 4

// ---------------------------------------------------------------------------
// Kernel 1: z_e [B, D, H*W] fp32  ->  z_bf16 [N, D] bf16 (tiled transpose)
// ---------------------------------------------------------------------------
__global__ __launch_bounds__(256) void vq_prep_z(const float* __restrict__ z,
                                                 bf16_t* __restrict__ zb) {
  __shared__ bf16_t tile[32][132];
  const int b  = blockIdx.x >> 6;        // 0..31
  const int dt = (blockIdx.x >> 3) & 7;  // 0..7   (d tile of 32)
  const int ht = blockIdx.x & 7;         // 0..7   (hw tile of 128)
  const int t  = threadIdx.x;

  const float* src = z + ((size_t)b * D_DIM + (size_t)dt * 32) * HW + (size_t)ht * 128;
#pragma unroll
  for (int i = 0; i < 16; ++i) {
    int idx = t + i * 256;
    int row = idx >> 7;                  // local d
    int col = idx & 127;                 // local hw
    tile[row][col] = (bf16_t)src[(size_t)row * HW + col];
  }
  __syncthreads();

  bf16_t* dst = zb + ((size_t)(b * HW + ht * 128)) * D_DIM + dt * 32;
#pragma unroll
  for (int i = 0; i < 16; ++i) {
    int idx = t + i * 256;
    int hwl = idx >> 5;                  // local hw
    int dl  = idx & 31;                  // local d
    dst[(size_t)hwl * D_DIM + dl] = tile[dl][hwl];
  }
}

// ---------------------------------------------------------------------------
// Kernel 2: emb [K, D] fp32 -> emb_bf16 [K, D], and emb_sq[k] = ||e_k||^2
// ---------------------------------------------------------------------------
__global__ __launch_bounds__(256) void vq_prep_emb(const float* __restrict__ emb,
                                                   bf16_t* __restrict__ eb,
                                                   float* __restrict__ esq) {
  const int k = blockIdx.x;
  const int d = threadIdx.x;
  float v = emb[(size_t)k * D_DIM + d];
  eb[(size_t)k * D_DIM + d] = (bf16_t)v;
  float s = v * v;
#pragma unroll
  for (int off = 16; off; off >>= 1) s += __shfl_xor(s, off, 32);
  __shared__ float wsum[8];
  if ((d & 31) == 0) wsum[d >> 5] = s;
  __syncthreads();
  if (d == 0) {
    float tot = 0.f;
#pragma unroll
    for (int i = 0; i < 8; ++i) tot += wsum[i];
    esq[k] = tot;
  }
}

// ---------------------------------------------------------------------------
// CDNA5 async copy: 32 bytes global -> LDS, no VGPR staging (ASYNCcnt).
// INST_OFFSET applies to both the LDS and global addresses (ISA 10. async).
// ---------------------------------------------------------------------------
__device__ __forceinline__ void async_copy_32B(unsigned lds_off, const bf16_t* g) {
  unsigned long long ga = (unsigned long long)(uintptr_t)g;
  asm volatile(
      "global_load_async_to_lds_b128 %0, %1, off\n\t"
      "global_load_async_to_lds_b128 %0, %1, off offset:16"
      :: "v"(lds_off), "v"(ga) : "memory");
}

// ---------------------------------------------------------------------------
// Kernel 3: fused GEMM + argmin.
// Block: 8 waves x 32 rows = 256 rows. Each wave hoists two A panels
// (2 x 16x256 bf16 = 128 VGPRs); B code-tiles (16 x 256 bf16 = 8KB) are
// double-buffered in LDS via async copies and shared by all 8 waves; each
// B tile feeds 2 WMMAs (M-tiling) -> half the B bytes per WMMA.
// ---------------------------------------------------------------------------
__global__ __launch_bounds__(256) void vq_argmin(const bf16_t* __restrict__ zb,
                                                 const bf16_t* __restrict__ eb,
                                                 const float* __restrict__ esq,
                                                 int* __restrict__ idx_out) {
  __shared__ bf16_t bufB[2][16 * BPITCH];   // 2 x 8.25 KB

  const int t      = threadIdx.x;
  const int wave   = t >> 5;
  const int lane   = t & 31;
  const int half   = lane >> 4;          // 0: lanes 0-15, 1: lanes 16-31
  const int l16    = lane & 15;
  const int n_base = blockIdx.x * 256 + wave * 32;

  // Staging role: 256 threads x 32B = one 8KB tile (16 codes x 512B).
  const int s_code  = t >> 4;            // 0..15
  const int s_chunk = t & 15;            // 0..15 (32B chunks per 512B row)
  const unsigned lds0 =
      (unsigned)(uintptr_t)(&bufB[0][0]) + (unsigned)(s_code * BPITCH + s_chunk * 16) * 2u;
  const unsigned lds1 =
      (unsigned)(uintptr_t)(&bufB[1][0]) + (unsigned)(s_code * BPITCH + s_chunk * 16) * 2u;
  const bf16_t* gstage = eb + (size_t)s_code * D_DIM + s_chunk * 16;

  // A layout (ISA 7.12.2, 16-bit A 16x32): lane row M=l16,
  //   regs 0-3: K = 8*half + 0..7 ; regs 4-7: K = 16 + 8*half + 0..7
  v16bf a0[8], a1[8];
  const bf16_t* zrow0 = zb + (size_t)(n_base + l16) * D_DIM;
  const bf16_t* zrow1 = zb + (size_t)(n_base + 16 + l16) * D_DIM;
#pragma unroll
  for (int db = 0; db < 8; ++db) {
    v8bf lo0 = *(const v8bf*)(zrow0 + db * 32 + 8 * half);
    v8bf hi0 = *(const v8bf*)(zrow0 + db * 32 + 16 + 8 * half);
    a0[db] = __builtin_shufflevector(lo0, hi0, 0, 1, 2, 3, 4, 5, 6, 7,
                                               8, 9, 10, 11, 12, 13, 14, 15);
    v8bf lo1 = *(const v8bf*)(zrow1 + db * 32 + 8 * half);
    v8bf hi1 = *(const v8bf*)(zrow1 + db * 32 + 16 + 8 * half);
    a1[db] = __builtin_shufflevector(lo1, hi1, 0, 1, 2, 3, 4, 5, 6, 7,
                                               8, 9, 10, 11, 12, 13, 14, 15);
  }

  float minv0[8], minv1[8];
  int   mini0[8], mini1[8];
#pragma unroll
  for (int r = 0; r < 8; ++r) {
    minv0[r] = 3.4e38f; mini0[r] = 0;
    minv1[r] = 3.4e38f; mini1[r] = 0;
  }

  // Prologue: stage tile 0.
  async_copy_32B(lds0, gstage);

  for (int kb = 0; kb < NKB; ++kb) {
    const int k = kb * 16 + l16;                 // this lane's code column
    const float ek = esq[k];

    // Issue next tile's async copies, then release only the current tile:
    // async ops complete in order, so waiting ASYNCcnt<=2 retires the 2
    // oldest (current-tile) copies while the 2 new ones stay in flight.
    if (kb + 1 < NKB) {
      async_copy_32B((kb & 1) ? lds0 : lds1, gstage + (size_t)(kb + 1) * 16 * D_DIM);
      asm volatile("s_wait_asynccnt 2" ::: "memory");
    } else {
      asm volatile("s_wait_asynccnt 0" ::: "memory");
    }
    __syncthreads();

    const bf16_t* brow = &bufB[kb & 1][0] + l16 * BPITCH + 8 * half;
    v8f acc0 = {}, acc1 = {};
#pragma unroll
    for (int db = 0; db < 8; ++db) {
      // B layout (ISA 7.12.2, 16-bit B 32x16): lane col N=l16,
      //   regs 0-3: K = 8*half + 0..7 ; regs 4-7: K = 16 + 8*half + 0..7
      v8bf blo = *(const v8bf*)(brow + db * 32);
      v8bf bhi = *(const v8bf*)(brow + db * 32 + 16);
      v16bf bm = __builtin_shufflevector(blo, bhi, 0, 1, 2, 3, 4, 5, 6, 7,
                                                   8, 9, 10, 11, 12, 13, 14, 15);
      acc0 = __builtin_amdgcn_wmma_f32_16x16x32_bf16(
          false, a0[db], false, bm, (short)0, acc0, false, false);
      acc1 = __builtin_amdgcn_wmma_f32_16x16x32_bf16(
          false, a1[db], false, bm, (short)0, acc1, false, false);
    }
    // C layout: VGPR r holds row M = 8*half + r, col = l16.
#pragma unroll
    for (int r = 0; r < 8; ++r) {
      float d0 = ek - 2.0f * acc0[r];
      if (d0 < minv0[r]) { minv0[r] = d0; mini0[r] = k; }
      float d1 = ek - 2.0f * acc1[r];
      if (d1 < minv1[r]) { minv1[r] = d1; mini1[r] = k; }
    }
    __syncthreads();   // all waves done reading buf[kb&1] before it is restaged
  }

  // Reduce over the 16 columns held in each 16-lane half (ties -> smaller idx)
#pragma unroll
  for (int r = 0; r < 8; ++r) {
    float v0 = minv0[r]; int i0 = mini0[r];
    float v1 = minv1[r]; int i1 = mini1[r];
#pragma unroll
    for (int off = 8; off; off >>= 1) {
      float ov0 = __shfl_xor(v0, off, 32);
      int   oi0 = __shfl_xor(i0, off, 32);
      if (ov0 < v0 || (ov0 == v0 && oi0 < i0)) { v0 = ov0; i0 = oi0; }
      float ov1 = __shfl_xor(v1, off, 32);
      int   oi1 = __shfl_xor(i1, off, 32);
      if (ov1 < v1 || (ov1 == v1 && oi1 < i1)) { v1 = ov1; i1 = oi1; }
    }
    mini0[r] = i0;
    mini1[r] = i1;
  }
  if (l16 == 0) {
    const int row0 = n_base + half * 8;
#pragma unroll
    for (int r = 0; r < 8; ++r) {
      idx_out[row0 + r]      = mini0[r];
      idx_out[row0 + 16 + r] = mini1[r];
    }
  }
}

// ---------------------------------------------------------------------------
// Kernel 4: gather z_q = emb[idx] in [B,D,H,W] order + per-block SSE partials.
// ---------------------------------------------------------------------------
__global__ __launch_bounds__(256) void vq_gather(const float* __restrict__ z,
                                                 const float* __restrict__ emb,
                                                 const int* __restrict__ idx,
                                                 float* __restrict__ out,
                                                 float* __restrict__ partial) {
  const int b = blockIdx.x >> 8;
  const int d = blockIdx.x & 255;
  const size_t base = ((size_t)b * D_DIM + d) * HW;
  float s = 0.f;
#pragma unroll
  for (int c = 0; c < 4; ++c) {
    int hw = c * 256 + threadIdx.x;
    int n  = b * HW + hw;
    int k  = idx[n];
    float v = emb[(size_t)k * D_DIM + d];   // gather: emb is L2-resident (1 MB)
    float df = v - z[base + hw];
    s += df * df;
    out[base + hw] = v;
  }
#pragma unroll
  for (int off = 16; off; off >>= 1) s += __shfl_xor(s, off, 32);
  __shared__ float wsum[8];
  if ((threadIdx.x & 31) == 0) wsum[threadIdx.x >> 5] = s;
  __syncthreads();
  if (threadIdx.x == 0) {
    float t = 0.f;
#pragma unroll
    for (int i = 0; i < 8; ++i) t += wsum[i];
    partial[blockIdx.x] = t;               // deterministic (no float atomics)
  }
}

// ---------------------------------------------------------------------------
// Kernel 5: reduce 8192 partials -> vq_loss = (1+BETA) * SSE / numel
// ---------------------------------------------------------------------------
__global__ __launch_bounds__(256) void vq_loss_final(const float* __restrict__ partial,
                                                     float* __restrict__ out_loss) {
  float s = 0.f;
  for (int i = threadIdx.x; i < 8192; i += 256) s += partial[i];
#pragma unroll
  for (int off = 16; off; off >>= 1) s += __shfl_xor(s, off, 32);
  __shared__ float wsum[8];
  if ((threadIdx.x & 31) == 0) wsum[threadIdx.x >> 5] = s;
  __syncthreads();
  if (threadIdx.x == 0) {
    float t = 0.f;
#pragma unroll
    for (int i = 0; i < 8; ++i) t += wsum[i];
    *out_loss = 1.25f * t / (float)NUMEL;  // loss_commit + BETA*loss_codebook
  }
}

// ---------------------------------------------------------------------------
extern "C" void kernel_launch(void* const* d_in, const int* in_sizes, int n_in,
                              void* d_out, int out_size, void* d_ws, size_t ws_size,
                              hipStream_t stream) {
  const float* z   = (const float*)d_in[0];   // [32, 256, 32, 32]
  const float* emb = (const float*)d_in[1];   // [1024, 256]
  float* out = (float*)d_out;                 // [numel z_q] + [1 loss]

  // Workspace carve (~17 MB total)
  char* ws = (char*)d_ws;
  bf16_t* zb  = (bf16_t*)ws;                                     // 16 MB
  ws += NUMEL * sizeof(bf16_t);
  bf16_t* eb  = (bf16_t*)ws;                                     // 512 KB
  ws += (size_t)K_CODES * D_DIM * sizeof(bf16_t);
  float* esq  = (float*)ws;                                      // 4 KB
  ws += (size_t)K_CODES * sizeof(float);
  int* idxb   = (int*)ws;                                        // 128 KB
  ws += (size_t)N_TOT * sizeof(int);
  float* part = (float*)ws;                                      // 32 KB

  vq_prep_z   <<<2048, 256, 0, stream>>>(z, zb);
  vq_prep_emb <<<K_CODES, 256, 0, stream>>>(emb, eb, esq);
  vq_argmin   <<<N_TOT / 256, 256, 0, stream>>>(zb, eb, esq, idxb);
  vq_gather   <<<B_SZ * D_DIM, 256, 0, stream>>>(z, emb, idxb, out, part);
  vq_loss_final<<<1, 256, 0, stream>>>(part, out + NUMEL);
}
